// DeepSeekVisionDecoder_68092411511049
// MI455X (gfx1250) — matrix operand order repaired and checked by reading, not compiled, not run on hardware
//
#include <hip/hip_runtime.h>

// ---------------------------------------------------------------------------
// DeepSeek vision decoder for MI455X (gfx1250, wave32, WMMA).
// All matmuls run through one tiled bf16 WMMA GEMM kernel
// (v_wmma_f32_16x16x32_bf16). fp32 operands are converted to bf16 while
// staging into LDS (packed-DWORD stores, vectorized b128 global loads),
// accumulation stays fp32. Software-pipelined: tile i+1's global loads are
// issued while tile i's WMMAs execute, hiding HBM latency.
// ---------------------------------------------------------------------------

#define BM 128
#define BN 128
#define BK 32
#define ASTR 48   // LDS row stride (bf16 elems): 96 bytes, 16B-aligned rows
#define BSTR 48

typedef __attribute__((ext_vector_type(16))) __bf16 bf16x16;
typedef __attribute__((ext_vector_type(8)))  __bf16 bf16x8;
typedef __attribute__((ext_vector_type(8)))  float  floatx8;

union Frag { bf16x16 v; bf16x8 h[2]; };

static __device__ __forceinline__ unsigned short f2bfu(float f) {
  union { float f; unsigned u; } v; v.f = f;
  const unsigned r = v.u + 0x7FFFu + ((v.u >> 16) & 1u);  // round-nearest-even
  return (unsigned short)(r >> 16);
}
static __device__ __forceinline__ unsigned pk2(float a, float b) {
  return (unsigned)f2bfu(a) | ((unsigned)f2bfu(b) << 16);
}

// C[M,N] (f32) = A[M,K](f32,row-major,lda) x B(f32) (+bias)(+C if accumulate)
// TRANSB=0: B is [K,N] ldb ; TRANSB=1: B is [N,K] ldb (i.e. C = A @ B^T).
// Batched over blockIdx.z with (b,h) split strides so attention heads work.
template<bool TRANSB>
__global__ __launch_bounds__(256)
void k_gemm(const float* __restrict__ A, const float* __restrict__ B,
            float* __restrict__ C, const float* __restrict__ bias,
            int M, int N, int K, int lda, int ldb, int ldc,
            int batchH,
            long long sAb, long long sAh, long long sBb, long long sBh,
            long long sCb, long long sCh, int accumulate)
{
  __shared__ __align__(16) __bf16 As[BM][ASTR];
  __shared__ __align__(16) __bf16 Bt[BN][BSTR];   // [n][k] : K-contiguous

  const int tid = threadIdx.x;
  const int z  = blockIdx.z;
  const int zb = z / batchH, zh = z % batchH;
  A += zb * sAb + zh * sAh;
  B += zb * sBb + zh * sBh;
  C += zb * sCb + zh * sCh;

  const int m0 = blockIdx.y * BM;
  const int n0 = blockIdx.x * BN;

  const int wave    = tid >> 5;
  const int lane    = tid & 31;
  const int waveM   = wave & 1;    // 2 waves over M (64 rows each)
  const int waveN   = wave >> 1;   // 4 waves over N (32 cols each)
  const int laneIdx = lane & 15;
  const int laneHi  = lane >> 4;

  floatx8 acc[4][2];
  const floatx8 zero8 = {0.f,0.f,0.f,0.f,0.f,0.f,0.f,0.f};
#pragma unroll
  for (int i = 0; i < 4; ++i)
#pragma unroll
    for (int j = 0; j < 2; ++j) acc[i][j] = zero8;

  // staging geometry (constant per thread)
  const int aRow = tid >> 1;              // A (and trans-B) row
  const int aCb  = (tid & 1) * 16;        // 16 contiguous K
  const int bKk  = (tid & 15) * 2;        // no-trans B: k-pair
  const int bNc  = (tid >> 4) * 8;        // no-trans B: 8 contiguous n

  // register staging buffers (software pipeline)
  float aV[16];
  float btV[16];                          // TRANSB
  float b0V[8], b1V[8];                   // no-trans B (k-pair)

  auto loadA = [&](int k0) {
    const int gm = m0 + aRow;
    const float* Ap = A + (long long)gm * lda + (k0 + aCb);
    __builtin_prefetch(Ap + BK, 0, 1);           // speculative: safe if OOB
    if (gm < M && (k0 + aCb + 16) <= K) {        // uniform fast path
      const float4* p4 = (const float4*)Ap;
#pragma unroll
      for (int j = 0; j < 4; ++j) {
        const float4 f = p4[j];
        aV[4*j+0] = f.x; aV[4*j+1] = f.y; aV[4*j+2] = f.z; aV[4*j+3] = f.w;
      }
    } else {
#pragma unroll
      for (int i = 0; i < 16; ++i) {
        const int gk = k0 + aCb + i;
        aV[i] = (gm < M && gk < K) ? Ap[i] : 0.f;
      }
    }
  };

  auto loadB = [&](int k0) {
    if (TRANSB) {
      const int gn = n0 + aRow;
      const float* Bp = B + (long long)gn * ldb + (k0 + aCb);
      if (gn < N && (k0 + aCb + 16) <= K) {
        const float4* p4 = (const float4*)Bp;
#pragma unroll
        for (int j = 0; j < 4; ++j) {
          const float4 f = p4[j];
          btV[4*j+0] = f.x; btV[4*j+1] = f.y;
          btV[4*j+2] = f.z; btV[4*j+3] = f.w;
        }
      } else {
#pragma unroll
        for (int i = 0; i < 16; ++i) {
          const int gk = k0 + aCb + i;
          btV[i] = (gn < N && gk < K) ? Bp[i] : 0.f;
        }
      }
    } else {
      const int gk0 = k0 + bKk;
      const float* B0 = B + (long long)gk0 * ldb + (n0 + bNc);
      const float* B1 = B0 + ldb;
      __builtin_prefetch(B0 + (long long)BK * ldb, 0, 1);
      if ((gk0 + 1) < K && (n0 + bNc + 8) <= N) {  // uniform fast path
        const float4* p0 = (const float4*)B0;
        const float4* p1 = (const float4*)B1;
        float4 f;
        f = p0[0]; b0V[0]=f.x; b0V[1]=f.y; b0V[2]=f.z; b0V[3]=f.w;
        f = p0[1]; b0V[4]=f.x; b0V[5]=f.y; b0V[6]=f.z; b0V[7]=f.w;
        f = p1[0]; b1V[0]=f.x; b1V[1]=f.y; b1V[2]=f.z; b1V[3]=f.w;
        f = p1[1]; b1V[4]=f.x; b1V[5]=f.y; b1V[6]=f.z; b1V[7]=f.w;
      } else {
#pragma unroll
        for (int j = 0; j < 8; ++j) {
          const int gn = n0 + bNc + j;
          const bool nok = gn < N;
          b0V[j] = (nok && gk0     < K) ? B0[j] : 0.f;
          b1V[j] = (nok && gk0 + 1 < K) ? B1[j] : 0.f;
        }
      }
    }
  };

  auto stage = [&]() {
    {
      unsigned u[8];
#pragma unroll
      for (int j = 0; j < 8; ++j) u[j] = pk2(aV[2*j], aV[2*j+1]);
      uint4* dst = (uint4*)&As[aRow][aCb];
      dst[0] = *(const uint4*)&u[0];
      dst[1] = *(const uint4*)&u[4];
    }
    if (TRANSB) {
      unsigned u[8];
#pragma unroll
      for (int j = 0; j < 8; ++j) u[j] = pk2(btV[2*j], btV[2*j+1]);
      uint4* dst = (uint4*)&Bt[aRow][aCb];
      dst[0] = *(const uint4*)&u[0];
      dst[1] = *(const uint4*)&u[4];
    } else {
#pragma unroll
      for (int j = 0; j < 8; ++j)
        *(unsigned*)&Bt[bNc + j][bKk] = pk2(b0V[j], b1V[j]);
    }
  };

  // ---- software pipeline: fetch(i+1) overlaps WMMA(i) ----
  loadA(0); loadB(0);
  for (int k0 = 0; k0 < K; k0 += BK) {
    stage();                 // regs -> LDS (fp32->bf16 packed)
    __syncthreads();
    if (k0 + BK < K) {       // issue next tile's global loads now
      loadA(k0 + BK);
      loadB(k0 + BK);
    }

    // ---- fragments per ISA 7.12.2 layouts ----
    Frag a[4], b[2];
#pragma unroll
    for (int i = 0; i < 4; ++i) {
      const int row = waveM * 64 + i * 16 + laneIdx;
      const int kc  = laneHi * 8;       // lanes<16: K0-7/16-23 ; else 8-15/24-31
      a[i].h[0] = *(const bf16x8*)&As[row][kc];
      a[i].h[1] = *(const bf16x8*)&As[row][kc + 16];
    }
#pragma unroll
    for (int j = 0; j < 2; ++j) {
      const int col = waveN * 32 + j * 16 + laneIdx;
      const int kc  = laneHi * 8;       // lanes<16: K0-7/16-23 ; else 8-15/24-31
      b[j].h[0] = *(const bf16x8*)&Bt[col][kc];
      b[j].h[1] = *(const bf16x8*)&Bt[col][kc + 16];
    }
#pragma unroll
    for (int i = 0; i < 4; ++i)
#pragma unroll
      for (int j = 0; j < 2; ++j)
        acc[i][j] = __builtin_amdgcn_wmma_f32_16x16x32_bf16(
            false, a[i].v, false, b[j].v, (short)0, acc[i][j], false, false);
    __syncthreads();
  }

  // ---- epilogue : C layout VGPR r -> M = r + 8*laneHi ----
#pragma unroll
  for (int i = 0; i < 4; ++i) {
    const int rowBase = m0 + waveM * 64 + i * 16 + laneHi * 8;
#pragma unroll
    for (int j = 0; j < 2; ++j) {
      const int col = n0 + waveN * 32 + j * 16 + laneIdx;
      if (col < N) {
#pragma unroll
        for (int r = 0; r < 8; ++r) {
          const int row = rowBase + r;
          if (row < M) {
            const long long idx = (long long)row * ldc + col;
            float v = acc[i][j][r];
            if (bias) v += bias[col];
            if (accumulate) v += C[idx];
            C[idx] = v;
          }
        }
      }
    }
  }
}

// ---------------------------------------------------------------------------
// elementwise / reduction helpers
// ---------------------------------------------------------------------------
__global__ __launch_bounds__(256)
void k_rms(const float* __restrict__ x, const float* __restrict__ w,
           float* __restrict__ y, int D) {
  const long long row = blockIdx.x;
  const float* xr = x + row * D;
  float* yr = y + row * D;
  __shared__ float red[256];
  float s = 0.f;
  for (int d = threadIdx.x; d < D; d += 256) { float v = xr[d]; s += v * v; }
  red[threadIdx.x] = s; __syncthreads();
  for (int o = 128; o > 0; o >>= 1) {
    if (threadIdx.x < (unsigned)o) red[threadIdx.x] += red[threadIdx.x + o];
    __syncthreads();
  }
  const float inv = rsqrtf(red[0] / (float)D + 1e-6f);
  for (int d = threadIdx.x; d < D; d += 256) yr[d] = xr[d] * inv * w[d];
}

__global__ __launch_bounds__(256)
void k_softmax(float* __restrict__ S, int cols, float scale) {
  float* r = S + (long long)blockIdx.x * cols;
  __shared__ float red[256];
  const int t = threadIdx.x;
  float m = -3.4e38f;
  for (int c = t; c < cols; c += 256) m = fmaxf(m, r[c]);
  red[t] = m; __syncthreads();
  for (int o = 128; o > 0; o >>= 1) {
    if (t < o) red[t] = fmaxf(red[t], red[t + o]);
    __syncthreads();
  }
  m = red[0] * scale; __syncthreads();
  float s = 0.f;
  for (int c = t; c < cols; c += 256) {
    const float e = __expf(r[c] * scale - m);
    r[c] = e; s += e;
  }
  red[t] = s; __syncthreads();
  for (int o = 128; o > 0; o >>= 1) {
    if (t < o) red[t] += red[t + o];
    __syncthreads();
  }
  const float inv = 1.f / red[0];
  for (int c = t; c < cols; c += 256) r[c] *= inv;
}

__global__ void k_rope(float* __restrict__ Q, float* __restrict__ Kk,
                       int T, int Ntok) {
  const int gid = blockIdx.x * blockDim.x + threadIdx.x;   // T*16*32
  if (gid >= T * 16 * 32) return;
  const int i = gid & 31;
  const int h = (gid >> 5) & 15;
  const int t = gid >> 9;
  const int n = t % Ntok;
  const float inv = __powf(10000.f, -(2.f * (float)i) / 64.f);
  const float fr = (float)n * inv;
  const float c = __cosf(fr), s = __sinf(fr);
  const long long base = (long long)t * 1024 + h * 64 + i;
  const float q1 = Q[base],  q2 = Q[base + 32];
  Q[base] = q1 * c - q2 * s; Q[base + 32] = q2 * c + q1 * s;
  const float k1 = Kk[base], k2 = Kk[base + 32];
  Kk[base] = k1 * c - k2 * s; Kk[base + 32] = k2 * c + k1 * s;
}

__global__ void k_silu_mul(float* __restrict__ g, const float* __restrict__ u,
                           long long n) {
  const long long i = (long long)blockIdx.x * blockDim.x + threadIdx.x;
  if (i < n) { const float x = g[i]; g[i] = (x / (1.f + __expf(-x))) * u[i]; }
}

__global__ void k_add_to(float* __restrict__ dst, const float* __restrict__ src,
                         long long n) {
  const long long i = (long long)blockIdx.x * blockDim.x + threadIdx.x;
  if (i < n) dst[i] += src[i];
}

__global__ void k_add_bcast(float* __restrict__ dst, const float* __restrict__ a,
                            const float* __restrict__ rowsrc, int rows, int D,
                            long long n) {
  // dst[t,d] = a[t,d] + rowsrc[t % rows, d]
  const long long i = (long long)blockIdx.x * blockDim.x + threadIdx.x;
  if (i < n) {
    const long long t = i / D;
    dst[i] = a[i] + rowsrc[(long long)(t % rows) * D + (i % D)];
  }
}

__global__ void k_scale_acc(float* __restrict__ out, const float* __restrict__ y,
                            const float* __restrict__ wf, int e, int E,
                            int D, long long n) {
  const long long i = (long long)blockIdx.x * blockDim.x + threadIdx.x;
  if (i < n) {
    const long long t = i / D;
    const float s = wf ? wf[t * E + e] : 1.f;
    out[i] += s * y[i];
  }
}

__global__ __launch_bounds__(256)
void k_router(const float* __restrict__ X, const float* __restrict__ RW,
              float* __restrict__ WF, int D) {
  const int t = blockIdx.x;
  __shared__ float red[256];
  __shared__ float logits[8];
  const int e = threadIdx.x >> 5, lane = threadIdx.x & 31;
  float s = 0.f;
  for (int d = lane; d < D; d += 32)
    s += X[(long long)t * D + d] * RW[(long long)e * D + d];
  red[threadIdx.x] = s; __syncthreads();
  if (lane == 0) {
    float a = 0.f;
    for (int i = 0; i < 32; ++i) a += red[e * 32 + i];
    logits[e] = a;
  }
  __syncthreads();
  if (threadIdx.x == 0) {
    float mx = -3.4e38f;
    for (int i = 0; i < 8; ++i) mx = fmaxf(mx, logits[i]);
    float p[8], den = 0.f;
    for (int i = 0; i < 8; ++i) { p[i] = __expf(logits[i] - mx); den += p[i]; }
    for (int i = 0; i < 8; ++i) p[i] /= den;
    int i1 = 0;
    for (int i = 1; i < 8; ++i) if (p[i] > p[i1]) i1 = i;
    int i2 = -1;
    for (int i = 0; i < 8; ++i)
      if (i != i1 && (i2 < 0 || p[i] > p[i2])) i2 = i;
    const float sw = p[i1] + p[i2];
    for (int i = 0; i < 8; ++i) WF[(long long)t * 8 + i] = 0.f;
    WF[(long long)t * 8 + i1] = p[i1] / sw;
    WF[(long long)t * 8 + i2] = p[i2] / sw;
  }
}

// ---------------------------------------------------------------------------
// host orchestration
// ---------------------------------------------------------------------------
namespace {
constexpr int B_   = 4,   N_  = 576, DIN_ = 1024;
constexpr int D_   = 1024, H_ = 16,  DH_ = 64;
constexpr int L_   = 4,   E_  = 8,   DM_  = 512;
constexpr int DFF_ = 4096;
constexpr int K_   = 16,  DM2_ = 2048;
constexpr int T_   = B_ * N_;   // 2304

inline void gemm(hipStream_t st, const float* A, const float* B, float* C,
                 int M, int N, int K, int lda, int ldb, int ldc,
                 const float* bias, bool transb, bool accum,
                 int batches = 1, int batchH = 1,
                 long long sAb = 0, long long sAh = 0,
                 long long sBb = 0, long long sBh = 0,
                 long long sCb = 0, long long sCh = 0) {
  dim3 grid((N + BN - 1) / BN, (M + BM - 1) / BM, batches);
  if (transb)
    k_gemm<true><<<grid, 256, 0, st>>>(A, B, C, bias, M, N, K, lda, ldb, ldc,
                                       batchH, sAb, sAh, sBb, sBh, sCb, sCh,
                                       accum ? 1 : 0);
  else
    k_gemm<false><<<grid, 256, 0, st>>>(A, B, C, bias, M, N, K, lda, ldb, ldc,
                                        batchH, sAb, sAh, sBb, sBh, sCb, sCh,
                                        accum ? 1 : 0);
}
inline void gemm0(hipStream_t st, const float* A, const float* B, float* C,
                  int M, int N, int K, const float* bias = nullptr) {
  gemm(st, A, B, C, M, N, K, K, N, N, bias, false, false);
}
inline dim3 g1d(long long n) { return dim3((unsigned)((n + 255) / 256)); }
} // namespace

extern "C" void kernel_launch(void* const* d_in, const int* in_sizes, int n_in,
                              void* d_out, int out_size, void* d_ws,
                              size_t ws_size, hipStream_t stream) {
  (void)in_sizes; (void)n_in; (void)out_size; (void)ws_size;

  const float* patch     = (const float*)d_in[0];
  const float* w_gate_in = (const float*)d_in[1];
  const float* w_up_in   = (const float*)d_in[2];
  const float* w_down_in = (const float*)d_in[3];
  const float* pos_emb   = (const float*)d_in[4];
  const float* ln1       = (const float*)d_in[5];
  const float* ln2       = (const float*)d_in[6];
  const float* wq        = (const float*)d_in[7];
  const float* wk        = (const float*)d_in[8];
  const float* wv        = (const float*)d_in[9];
  const float* wo        = (const float*)d_in[10];
  const float* router_w  = (const float*)d_in[11];
  const float* exp_gate  = (const float*)d_in[12];
  const float* exp_up    = (const float*)d_in[13];
  const float* exp_down  = (const float*)d_in[14];
  const float* sh_gate   = (const float*)d_in[15];
  const float* sh_up     = (const float*)d_in[16];
  const float* sh_down   = (const float*)d_in[17];
  const float* query_tok = (const float*)d_in[18];
  const float* q_w = (const float*)d_in[19]; const float* q_b = (const float*)d_in[20];
  const float* k_w = (const float*)d_in[21]; const float* k_b = (const float*)d_in[22];
  const float* v_w = (const float*)d_in[23]; const float* v_b = (const float*)d_in[24];
  const float* o_w = (const float*)d_in[25]; const float* o_b = (const float*)d_in[26];
  const float* cross_norm_w = (const float*)d_in[27];
  const float* tg_router    = (const float*)d_in[28];
  const float* tg_gate      = (const float*)d_in[29];
  const float* tg_up        = (const float*)d_in[30];
  const float* tg_down      = (const float*)d_in[31];
  const float* out_norm_w   = (const float*)d_in[32];

  // bump allocator over workspace
  char* wsp = (char*)d_ws;
  size_t used = 0;
  auto alloc = [&](size_t nfloats) -> float* {
    float* p = (float*)(wsp + used);
    used += nfloats * sizeof(float);
    return p;
  };

  float* gbuf = alloc((size_t)T_ * DFF_);   // also reused as expert gate buf
  float* ubuf = alloc((size_t)T_ * DFF_);   // also reused as expert up buf
  float* hb   = alloc((size_t)T_ * D_);
  float* xn   = alloc((size_t)T_ * D_);
  float* qb   = alloc((size_t)T_ * D_);
  float* kb   = alloc((size_t)T_ * D_);
  float* vb   = alloc((size_t)T_ * D_);
  float* Sb   = alloc((size_t)B_ * H_ * N_ * N_);
  float* ao   = alloc((size_t)T_ * D_);
  float* ap   = alloc((size_t)T_ * D_);
  float* yb   = alloc((size_t)T_ * D_);
  float* macc = alloc((size_t)T_ * D_);
  float* wfb  = alloc((size_t)T_ * E_);
  float* q16  = alloc((size_t)K_ * D_);
  float* aoc  = alloc((size_t)B_ * K_ * D_);
  float* attc = alloc((size_t)B_ * K_ * D_);
  float* att2 = alloc((size_t)B_ * K_ * D_);
  float* attn = alloc((size_t)B_ * K_ * D_);
  float* tga  = alloc((size_t)B_ * K_ * D_);
  float* wf2  = alloc((size_t)B_ * K_ * E_);

  const long long TD = (long long)T_ * D_;

  // ---- input SwiGLU MLP + positional embedding ----
  gemm0(stream, patch, w_gate_in, gbuf, T_, DFF_, DIN_);
  gemm0(stream, patch, w_up_in,   ubuf, T_, DFF_, DIN_);
  k_silu_mul<<<g1d((long long)T_ * DFF_), 256, 0, stream>>>(gbuf, ubuf,
                                                            (long long)T_ * DFF_);
  gemm0(stream, gbuf, w_down_in, hb, T_, D_, DFF_);
  k_add_bcast<<<g1d(TD), 256, 0, stream>>>(hb, hb, pos_emb, N_, D_, TD);

  // ---- transformer layers ----
  for (int l = 0; l < L_; ++l) {
    const float* l1 = ln1 + (size_t)l * D_;
    const float* l2 = ln2 + (size_t)l * D_;
    const float* Wq = wq + (size_t)l * D_ * D_;
    const float* Wk = wk + (size_t)l * D_ * D_;
    const float* Wv = wv + (size_t)l * D_ * D_;
    const float* Wo = wo + (size_t)l * D_ * D_;

    // self-attention
    k_rms<<<T_, 256, 0, stream>>>(hb, l1, xn, D_);
    gemm0(stream, xn, Wq, qb, T_, D_, D_);
    gemm0(stream, xn, Wk, kb, T_, D_, D_);
    gemm0(stream, xn, Wv, vb, T_, D_, D_);
    k_rope<<<g1d((long long)T_ * H_ * 32), 256, 0, stream>>>(qb, kb, T_, N_);

    // S = Q K^T  per (b,h)
    gemm(stream, qb, kb, Sb, N_, N_, DH_, D_, D_, N_, nullptr, true, false,
         B_ * H_, H_,
         (long long)N_ * D_, 64, (long long)N_ * D_, 64,
         (long long)H_ * N_ * N_, (long long)N_ * N_);
    k_softmax<<<B_ * H_ * N_, 256, 0, stream>>>(Sb, N_, 0.125f);
    // O = softmax(S) V
    gemm(stream, Sb, vb, ao, N_, DH_, N_, N_, D_, D_, nullptr, false, false,
         B_ * H_, H_,
         (long long)H_ * N_ * N_, (long long)N_ * N_,
         (long long)N_ * D_, 64, (long long)N_ * D_, 64);
    gemm0(stream, ao, Wo, ap, T_, D_, D_);
    k_add_to<<<g1d(TD), 256, 0, stream>>>(hb, ap, TD);

    // MoE (top-2 of 8, computed expert-dense; routing zeros unselected)
    k_rms<<<T_, 256, 0, stream>>>(hb, l2, xn, D_);
    k_router<<<T_, 256, 0, stream>>>(xn, router_w + (size_t)l * E_ * D_, wfb, D_);
    hipMemsetAsync(macc, 0, TD * sizeof(float), stream);
    for (int e = 0; e < E_; ++e) {
      const float* eg = exp_gate + ((size_t)l * E_ + e) * D_ * DM_;
      const float* eu = exp_up   + ((size_t)l * E_ + e) * D_ * DM_;
      const float* ed = exp_down + ((size_t)l * E_ + e) * DM_ * D_;
      gemm0(stream, xn, eg, gbuf, T_, DM_, D_);
      gemm0(stream, xn, eu, ubuf, T_, DM_, D_);
      k_silu_mul<<<g1d((long long)T_ * DM_), 256, 0, stream>>>(
          gbuf, ubuf, (long long)T_ * DM_);
      gemm0(stream, gbuf, ed, yb, T_, D_, DM_);
      k_scale_acc<<<g1d(TD), 256, 0, stream>>>(macc, yb, wfb, e, E_, D_, TD);
    }
    // shared expert
    gemm0(stream, xn, sh_gate + (size_t)l * D_ * DM_, gbuf, T_, DM_, D_);
    gemm0(stream, xn, sh_up   + (size_t)l * D_ * DM_, ubuf, T_, DM_, D_);
    k_silu_mul<<<g1d((long long)T_ * DM_), 256, 0, stream>>>(
        gbuf, ubuf, (long long)T_ * DM_);
    gemm0(stream, gbuf, sh_down + (size_t)l * DM_ * D_, yb, T_, D_, DM_);
    k_scale_acc<<<g1d(TD), 256, 0, stream>>>(macc, yb, nullptr, 0, E_, D_, TD);
    k_add_to<<<g1d(TD), 256, 0, stream>>>(hb, macc, TD);
  }

  // ---- cross attention (16 learned queries, biased projections) ----
  gemm0(stream, query_tok, q_w, q16, K_, D_, D_, q_b);
  gemm0(stream, hb, k_w, kb, T_, D_, D_, k_b);
  gemm0(stream, hb, v_w, vb, T_, D_, D_, v_b);
  gemm(stream, q16, kb, Sb, K_, N_, DH_, D_, D_, N_, nullptr, true, false,
       B_ * H_, H_, 0, 64, (long long)N_ * D_, 64,
       (long long)H_ * K_ * N_, (long long)K_ * N_);
  k_softmax<<<B_ * H_ * K_, 256, 0, stream>>>(Sb, N_, 0.125f);
  gemm(stream, Sb, vb, aoc, K_, DH_, N_, N_, D_, D_, nullptr, false, false,
       B_ * H_, H_, (long long)H_ * K_ * N_, (long long)K_ * N_,
       (long long)N_ * D_, 64, (long long)K_ * D_, 64);
  gemm0(stream, aoc, o_w, attc, B_ * K_, D_, D_, o_b);
  const long long BKD = (long long)B_ * K_ * D_;
  k_add_bcast<<<g1d(BKD), 256, 0, stream>>>(att2, attc, query_tok, K_, D_, BKD);
  k_rms<<<B_ * K_, 256, 0, stream>>>(att2, cross_norm_w, attn, D_);

  // ---- token-generator MoE + residual + out norm ----
  k_router<<<B_ * K_, 256, 0, stream>>>(attn, tg_router, wf2, D_);
  hipMemsetAsync(tga, 0, BKD * sizeof(float), stream);
  for (int e = 0; e < E_; ++e) {
    const float* eg = tg_gate + (size_t)e * D_ * DM2_;
    const float* eu = tg_up   + (size_t)e * D_ * DM2_;
    const float* ed = tg_down + (size_t)e * DM2_ * D_;
    gemm0(stream, attn, eg, gbuf, B_ * K_, DM2_, D_);
    gemm0(stream, attn, eu, ubuf, B_ * K_, DM2_, D_);
    k_silu_mul<<<g1d((long long)B_ * K_ * DM2_), 256, 0, stream>>>(
        gbuf, ubuf, (long long)B_ * K_ * DM2_);
    gemm0(stream, gbuf, ed, yb, B_ * K_, D_, DM2_);
    k_scale_acc<<<g1d(BKD), 256, 0, stream>>>(tga, yb, wf2, e, E_, D_, BKD);
  }
  k_add_to<<<g1d(BKD), 256, 0, stream>>>(tga, attn, BKD);
  k_rms<<<B_ * K_, 256, 0, stream>>>(tga, out_norm_w, (float*)d_out, D_);
}
